// GCN_17841294147604
// MI455X (gfx1250) — matrix operand order repaired and checked by reading, not compiled, not run on hardware
//
#include <hip/hip_runtime.h>

#define NN   50000
#define NE   800000
#define GG   64
#define HD   128
#define HD2  64
#define HD3  32
#define OUTD 2
#define EPSV 1e-5f

typedef __attribute__((ext_vector_type(16))) __bf16 v16bf;
typedef __attribute__((ext_vector_type(8)))  float  v8f;

__device__ __forceinline__ unsigned short f2bf(float f) {
  unsigned int u = __float_as_uint(f);
  u += 0x7FFFu + ((u >> 16) & 1u);          // round to nearest even
  return (unsigned short)(u >> 16);
}

// Native hardware fp32 atomic add (GLOBAL_ATOMIC_ADD_F32), no CAS expansion.
__device__ __forceinline__ void atomAddF(float* p, float v) {
  unsafeAtomicAdd(p, v);
}

// ---------------- small utility kernels ----------------
__global__ void k_fill(float* p, float v, int n) {
  int i = blockIdx.x * blockDim.x + threadIdx.x;
  if (i < n) p[i] = v;
}

__global__ void k_deg(const int* __restrict__ dst, const float* __restrict__ w,
                      float* __restrict__ deg, int e) {
  int i = blockIdx.x * blockDim.x + threadIdx.x;
  if (i < e) atomAddF(&deg[dst[i]], w[i]);
}

__global__ void k_dinv(float* deg, int n) {
  int i = blockIdx.x * blockDim.x + threadIdx.x;
  if (i < n) { float d = deg[i]; deg[i] = d > 0.f ? rsqrtf(d) : 0.f; }
}

__global__ void k_norm(const int* __restrict__ src, const int* __restrict__ dst,
                       const float* __restrict__ w, const float* __restrict__ dinv,
                       float* __restrict__ nrm, int e) {
  int i = blockIdx.x * blockDim.x + threadIdx.x;
  if (i < e) nrm[i] = dinv[src[i]] * w[i] * dinv[dst[i]];
}

// ---------------- fragment prep ----------------
// B-fragment order for W (128x128, row-major K x N):
//   wfrag[((cb*4+kb)*32 + lane)*16 + t] = bf16( W[(kb*32 + (lane>>4)*8 + t + (t&8))*128 + cb*16 + (lane&15)] )
__global__ void k_wprep(const float* __restrict__ W, unsigned short* __restrict__ frag) {
  int idx = blockIdx.x * blockDim.x + threadIdx.x;       // 8*4*32*16 = 16384
  if (idx >= 16384) return;
  int t    = idx & 15;
  int lane = (idx >> 4) & 31;
  int kb   = (idx >> 9) & 3;
  int cb   = idx >> 11;
  int k = kb * 32 + ((lane >> 4) << 3) + t + (t & 8);
  int n = cb * 16 + (lane & 15);
  frag[idx] = f2bf(W[k * HD + n]);
}

// A-fragment order for activations (N x 128, row-major):
//   afrag[((rb*4+kb)*32 + lane)*16 + t] = bf16( X[(rb*16 + (lane&15))*128 + kb*32 + (lane>>4)*8 + t + (t&8)] )
// (ISA 16-bit A layout: lane<16 holds K 0-7 & 16-23; lane>=16 holds K 8-15 & 24-31 of each 32-K block)
__global__ void k_aprep(const float* __restrict__ X, unsigned short* __restrict__ afrag) {
  int idx = blockIdx.x * blockDim.x + threadIdx.x;       // NN*128
  if (idx >= NN * HD) return;
  int t    = idx & 15;
  int lane = (idx >> 4) & 31;
  int kb   = (idx >> 9) & 3;
  int rb   = idx >> 11;
  int row = rb * 16 + (lane & 15);
  int k   = kb * 32 + ((lane >> 4) << 3) + t + (t & 8);
  afrag[idx] = f2bf(X[(size_t)row * HD + k]);
}

// ---------------- WMMA GEMM: T = Hin(N x 128) @ W(128 x 128) ----------------
// Pure fragment loads: 4x (32B A + 32B B) + 4 WMMA per 16x16 tile.
__global__ void k_gemm(const unsigned short* __restrict__ afrag,
                       const unsigned short* __restrict__ wfrag,
                       float* __restrict__ T) {
  int lane = threadIdx.x & 31;
  int wave = threadIdx.x >> 5;           // column block 0..7
  int rb   = blockIdx.x;                 // row block
  v8f acc = {0.f, 0.f, 0.f, 0.f, 0.f, 0.f, 0.f, 0.f};
#pragma unroll
  for (int kb = 0; kb < 4; ++kb) {
    v16bf a = *(const v16bf*)(afrag + (((size_t)(rb * 4 + kb) * 32 + lane) << 4));
    v16bf b = *(const v16bf*)(wfrag + (((wave * 4 + kb) * 32 + lane) << 4));
    acc = __builtin_amdgcn_wmma_f32_16x16x32_bf16(false, a, false, b,
                                                  (short)0, acc, false, false);
  }
  int col = wave * 16 + (lane & 15);
  int m0  = rb * 16 + ((lane >> 4) << 3);
#pragma unroll
  for (int j = 0; j < 8; ++j)
    T[(size_t)(m0 + j) * HD + col] = acc[j];
}

// ---------------- aggregation ----------------
// out[i][c] = dinv[i]^2 * T[i][c] + bias[c]   (self-loop term; initializes out)
__global__ void k_selfloop(const float* __restrict__ T, const float* __restrict__ dinv,
                           const float* __restrict__ bias, float* __restrict__ out) {
  int idx = blockIdx.x * blockDim.x + threadIdx.x;
  if (idx >= NN * 32) return;
  int node = idx >> 5;
  int c4   = (idx & 31) << 2;
  float s = dinv[node]; s = s * s;
  float4 t = *(const float4*)(T + (size_t)node * HD + c4);
  float4 o;
  o.x = s * t.x + bias[c4 + 0];
  o.y = s * t.y + bias[c4 + 1];
  o.z = s * t.z + bias[c4 + 2];
  o.w = s * t.w + bias[c4 + 3];
  *(float4*)(out + (size_t)node * HD + c4) = o;
}

// one wave per edge: lane handles 4 channels
__global__ void k_edges(const int* __restrict__ src, const int* __restrict__ dst,
                        const float* __restrict__ nrm, const float* __restrict__ T,
                        float* __restrict__ out) {
  int idx  = blockIdx.x * blockDim.x + threadIdx.x;
  int edge = idx >> 5;
  if (edge >= NE) return;
  int lane = idx & 31;
  int s = src[edge], d = dst[edge];
  float w = nrm[edge];
  int c4 = lane << 2;
  float4 t = *(const float4*)(T + (size_t)s * HD + c4);
  float* op = out + (size_t)d * HD + c4;
  atomAddF(op + 0, w * t.x);
  atomAddF(op + 1, w * t.y);
  atomAddF(op + 2, w * t.z);
  atomAddF(op + 3, w * t.w);
}

// ---------------- batch norm ----------------
__global__ void k_bnstats(const float* __restrict__ h, float* __restrict__ stats, int n) {
  __shared__ float ss[256], sq[256];
  int c = threadIdx.x & 127;
  int half = threadIdx.x >> 7;
  int row0 = blockIdx.x * 128;
  float s = 0.f, q = 0.f;
  for (int r = row0 + half; r < row0 + 128 && r < n; r += 2) {
    float v = h[(size_t)r * HD + c];
    s += v; q += v * v;
  }
  ss[threadIdx.x] = s; sq[threadIdx.x] = q;
  __syncthreads();
  if (half == 0) {
    atomAddF(&stats[c],       s + ss[threadIdx.x + 128]);
    atomAddF(&stats[128 + c], q + sq[threadIdx.x + 128]);
  }
}

__global__ void k_bnapply(float* __restrict__ h, const float* __restrict__ stats,
                          const float* __restrict__ g, const float* __restrict__ be) {
  int idx = blockIdx.x * blockDim.x + threadIdx.x;
  if (idx >= NN * 32) return;
  int node = idx >> 5;
  int c4   = (idx & 31) << 2;
  const float invn = 1.0f / (float)NN;
  float* hp = h + (size_t)node * HD + c4;
  float4 v = *(const float4*)hp;
  float r[4] = {v.x, v.y, v.z, v.w};
#pragma unroll
  for (int k = 0; k < 4; ++k) {
    int c = c4 + k;
    float mu  = stats[c] * invn;
    float var = stats[128 + c] * invn - mu * mu;
    float inv = rsqrtf(var + EPSV);
    float z = (r[k] - mu) * inv * g[c] + be[c];
    r[k] = z > 0.f ? z : 0.f;
  }
  float4 o = {r[0], r[1], r[2], r[3]};
  *(float4*)hp = o;
}

// ---------------- pooling ----------------
__global__ void k_counts(const int* __restrict__ batch, float* __restrict__ counts, int n) {
  int i = blockIdx.x * blockDim.x + threadIdx.x;
  if (i < n) atomAddF(&counts[batch[i]], 1.0f);
}

__global__ void k_poolsum(const float* __restrict__ h, const int* __restrict__ batch,
                          float* __restrict__ pooled) {
  int idx = blockIdx.x * blockDim.x + threadIdx.x;
  if (idx >= NN * 32) return;
  int node = idx >> 5;
  int c4   = (idx & 31) << 2;
  int b = batch[node];
  float4 v = *(const float4*)(h + (size_t)node * HD + c4);
  float* pp = pooled + (size_t)b * HD + c4;
  atomAddF(pp + 0, v.x);
  atomAddF(pp + 1, v.y);
  atomAddF(pp + 2, v.z);
  atomAddF(pp + 3, v.w);
}

// ---------------- MLP head (single block) ----------------
__global__ void k_head(const float* __restrict__ pooled, const float* __restrict__ counts,
                       const float* __restrict__ Wm1, const float* __restrict__ bm1,
                       const float* __restrict__ Wm2, const float* __restrict__ bm2,
                       const float* __restrict__ Wm3, const float* __restrict__ bm3,
                       const float* __restrict__ Wm4, const float* __restrict__ bm4,
                       float* __restrict__ out) {
  __shared__ float smem[16384];          // 64 KB
  float* p  = smem;                      // 64 x 128
  float* z1 = smem + 8192;               // 64 x 128
  int t = threadIdx.x;
  for (int i = t; i < GG * HD; i += 256) {
    int g = i >> 7;
    float c = counts[g]; c = c < 1.f ? 1.f : c;
    p[i] = pooled[i] / c;
  }
  __syncthreads();
  for (int i = t; i < GG * HD; i += 256) {          // 128 -> 128
    int g = i >> 7, j = i & 127;
    float s = bm1[j];
    for (int k = 0; k < HD; ++k) s += p[g * HD + k] * Wm1[k * HD + j];
    z1[i] = s > 0.f ? s : 0.f;
  }
  __syncthreads();
  float* z2 = smem;                      // 64 x 64 (overwrites p)
  for (int i = t; i < GG * HD2; i += 256) {          // 128 -> 64
    int g = i >> 6, j = i & 63;
    float s = bm2[j];
    for (int k = 0; k < HD; ++k) s += z1[g * HD + k] * Wm2[k * HD2 + j];
    z2[i] = s > 0.f ? s : 0.f;
  }
  __syncthreads();
  float* z3 = smem + 4096;               // 64 x 32
  for (int i = t; i < GG * HD3; i += 256) {          // 64 -> 32
    int g = i >> 5, j = i & 31;
    float s = bm3[j];
    for (int k = 0; k < HD2; ++k) s += z2[g * HD2 + k] * Wm3[k * HD3 + j];
    z3[i] = s > 0.f ? s : 0.f;
  }
  __syncthreads();
  for (int i = t; i < GG * OUTD; i += 256) {         // 32 -> 2
    int g = i >> 1, j = i & 1;
    float s = bm4[j];
    for (int k = 0; k < HD3; ++k) s += z3[g * HD3 + k] * Wm4[k * OUTD + j];
    out[i] = s;
  }
}

// ---------------- driver ----------------
extern "C" void kernel_launch(void* const* d_in, const int* in_sizes, int n_in,
                              void* d_out, int out_size, void* d_ws, size_t ws_size,
                              hipStream_t stream) {
  (void)in_sizes; (void)n_in; (void)out_size; (void)ws_size;
  const float* x    = (const float*)d_in[0];
  const int*   esrc = (const int*)d_in[1];
  const int*   edst = (const int*)d_in[2];
  const float* ew   = (const float*)d_in[3];
  const int*   batch= (const int*)d_in[4];
  const float* W[3]  = {(const float*)d_in[5], (const float*)d_in[7], (const float*)d_in[9]};
  const float* bb[3] = {(const float*)d_in[6], (const float*)d_in[8], (const float*)d_in[10]};
  const float* gg[3] = {(const float*)d_in[11], (const float*)d_in[13], (const float*)d_in[15]};
  const float* be[3] = {(const float*)d_in[12], (const float*)d_in[14], (const float*)d_in[16]};

  float* ws = (float*)d_ws;
  // workspace layout (floats): deg/dinv | norm | bufA | bufB | afrag | wfrag | stats | pooled | counts
  float*          deg    = ws;                               // NN
  float*          nrm    = ws + 50176;                       // NE
  float*          bufA   = ws + 850176;                      // NN*128
  float*          bufB   = ws + 7250176;                     // NN*128
  unsigned short* afrag  = (unsigned short*)(ws + 13650176); // NN*128 bf16 (3.2M floats)
  unsigned short* wfrag  = (unsigned short*)(ws + 16850176); // 16384 bf16
  float*          stats  = ws + 16858368;                    // 256
  float*          pooled = ws + 16858624;                    // 8192
  float*          counts = ws + 16866816;                    // 64

  const int B = 256;
  k_fill<<<(NN + B - 1) / B, B, 0, stream>>>(deg, 1.0f, NN);   // self-loop weight
  k_deg <<<(NE + B - 1) / B, B, 0, stream>>>(edst, ew, deg, NE);
  k_dinv<<<(NN + B - 1) / B, B, 0, stream>>>(deg, NN);         // deg -> dinv in place
  k_norm<<<(NE + B - 1) / B, B, 0, stream>>>(esrc, edst, ew, deg, nrm, NE);

  for (int L = 0; L < 3; ++L) {
    const float* hin = (L == 0) ? x : bufB;
    k_wprep   <<<16384 / B, B, 0, stream>>>(W[L], wfrag);
    k_aprep   <<<(NN * HD + B - 1) / B, B, 0, stream>>>(hin, afrag);
    k_gemm    <<<NN / 16, B, 0, stream>>>(afrag, wfrag, bufA);
    k_selfloop<<<(NN * 32 + B - 1) / B, B, 0, stream>>>(bufA, deg, bb[L], bufB);
    k_edges   <<<(NE * 32 + B - 1) / B, B, 0, stream>>>(esrc, edst, nrm, bufA, bufB);
    k_fill    <<<1, B, 0, stream>>>(stats, 0.f, 256);
    k_bnstats <<<(NN + 127) / 128, B, 0, stream>>>(bufB, stats, NN);
    k_bnapply <<<(NN * 32 + B - 1) / B, B, 0, stream>>>(bufB, stats, gg[L], be[L]);
  }

  k_fill   <<<(8192 + B - 1) / B, B, 0, stream>>>(pooled, 0.f, 8192);
  k_fill   <<<1, 64, 0, stream>>>(counts, 0.f, 64);
  k_counts <<<(NN + B - 1) / B, B, 0, stream>>>(batch, counts, NN);
  k_poolsum<<<(NN * 32 + B - 1) / B, B, 0, stream>>>(bufB, batch, pooled);

  k_head<<<1, B, 0, stream>>>(pooled, counts,
                              (const float*)d_in[17], (const float*)d_in[18],
                              (const float*)d_in[19], (const float*)d_in[20],
                              (const float*)d_in[21], (const float*)d_in[22],
                              (const float*)d_in[23], (const float*)d_in[24],
                              (float*)d_out);
}
